// SGFormer_25228637897373
// MI455X (gfx1250) — matrix operand order repaired and checked by reading, not compiled, not run on hardware
//
#include <hip/hip_runtime.h>
#include <hip/hip_bf16.h>

// ---------------------------------------------------------------------------
// SGFormer forward on gfx1250 (MI455X): WMMA bf16 GEMM pipeline + GCN scatter
// ---------------------------------------------------------------------------

typedef __attribute__((ext_vector_type(16))) __bf16 v16bf;
typedef __attribute__((ext_vector_type(8)))  __bf16 v8bf;
typedef __attribute__((ext_vector_type(4)))  __bf16 v4bf;
typedef __attribute__((ext_vector_type(8)))  float  v8f;

#define TB 256          // threads per block (8 waves of 32)
#define BM 128          // block tile M
#define BN 128          // block tile N
#define BK 32           // K step == WMMA K
#define LDK 40          // padded K stride in LDS (bf16 elems; 80B rows, 16B aligned)

// ACT: 0 = none, 1 = relu, 2 = elu(x)+1
template<int ACT, bool TRANSA>
__global__ __launch_bounds__(TB)
void gemm_bf16_wmma(const float* __restrict__ A, const float* __restrict__ B,
                    float* __restrict__ C, int M, int N, int K,
                    int lda, int ldb, int ldc,
                    const float* __restrict__ bias,
                    const float* __restrict__ rowscale, int beta)
{
    __shared__ __bf16 As[BM][LDK];
    __shared__ __bf16 Bs[BN][LDK];         // stored transposed: [n][k]

    const int tid  = threadIdx.x;
    const int lane = tid & 31;
    const int wid  = tid >> 5;
    const int l16  = lane & 15;
    const int lhi  = (lane >> 4) & 1;      // lane group 0/1
    const int wm   = (wid & 1) << 6;       // wave M offset: 0 or 64
    const int wn   = (wid >> 1) << 5;      // wave N offset: 0,32,64,96

    const int m0 = blockIdx.y * BM;
    const int n0 = blockIdx.x * BN;

    // split-K via gridDim.z (used for k^T v); chunks are multiples of BK
    const int nchunk = (int)gridDim.z;
    int kper   = ((K + nchunk * BK - 1) / (nchunk * BK)) * BK;
    int kstart = blockIdx.z * kper;
    int kend   = min(kstart + kper, K);

    v8f acc[4][2] = {};

    const bool mfull = (m0 + BM) <= M;

    for (int k0 = kstart; k0 < kend; k0 += BK) {
        const bool kfull = (k0 + BK) <= kend;
        // ---- stage A tile (BM x BK) as bf16 ----
        if (!TRANSA) {
            if (mfull && kfull) {
                #pragma unroll
                for (int t = 0; t < 4; ++t) {
                    int linear = tid + t * TB;          // 1024 float4 slots
                    int row = linear >> 3;              // 0..127
                    int c4  = (linear & 7) << 2;        // 0,4,..28
                    float4 val = *(const float4*)(A + (size_t)(m0 + row) * lda + k0 + c4);
                    v4bf p;
                    p[0] = (__bf16)val.x; p[1] = (__bf16)val.y;
                    p[2] = (__bf16)val.z; p[3] = (__bf16)val.w;
                    *(v4bf*)&As[row][c4] = p;
                }
            } else {
                #pragma unroll
                for (int t = 0; t < 4; ++t) {
                    int linear = tid + t * TB;
                    int row = linear >> 3;
                    int c4  = (linear & 7) << 2;
                    float4 val = make_float4(0.f, 0.f, 0.f, 0.f);
                    if ((m0 + row) < M && (k0 + c4 + 4) <= kend)
                        val = *(const float4*)(A + (size_t)(m0 + row) * lda + k0 + c4);
                    v4bf p;
                    p[0] = (__bf16)val.x; p[1] = (__bf16)val.y;
                    p[2] = (__bf16)val.z; p[3] = (__bf16)val.w;
                    *(v4bf*)&As[row][c4] = p;
                }
            }
        } else {
            // logical A[m,k] = G[k*lda + m]  (G is K x M row-major)
            if (mfull && kfull) {
                #pragma unroll
                for (int t = 0; t < 4; ++t) {
                    int linear = tid + t * TB;          // 1024 float4 slots
                    int kk = linear >> 5;               // 0..31
                    int mm = (linear & 31) << 2;        // 0,4,..124
                    float4 val = *(const float4*)(A + (size_t)(k0 + kk) * lda + m0 + mm);
                    As[mm + 0][kk] = (__bf16)val.x;
                    As[mm + 1][kk] = (__bf16)val.y;
                    As[mm + 2][kk] = (__bf16)val.z;
                    As[mm + 3][kk] = (__bf16)val.w;
                }
            } else {
                #pragma unroll
                for (int t = 0; t < 4; ++t) {
                    int linear = tid + t * TB;
                    int kk = linear >> 5;
                    int mm = (linear & 31) << 2;
                    float4 val = make_float4(0.f, 0.f, 0.f, 0.f);
                    int gk = k0 + kk;
                    if (gk < kend && (m0 + mm + 4) <= M)
                        val = *(const float4*)(A + (size_t)gk * lda + m0 + mm);
                    As[mm + 0][kk] = (__bf16)val.x;
                    As[mm + 1][kk] = (__bf16)val.y;
                    As[mm + 2][kk] = (__bf16)val.z;
                    As[mm + 3][kk] = (__bf16)val.w;
                }
            }
        }
        // ---- stage B tile (BK x BN) transposed into Bs[n][k] ----
        if (kfull) {
            #pragma unroll
            for (int t = 0; t < 4; ++t) {
                int linear = tid + t * TB;              // 1024 float4 slots
                int kk = linear >> 5;                   // 0..31
                int nn = (linear & 31) << 2;            // 0,4,..124
                float4 val = *(const float4*)(B + (size_t)(k0 + kk) * ldb + n0 + nn);
                Bs[nn + 0][kk] = (__bf16)val.x;
                Bs[nn + 1][kk] = (__bf16)val.y;
                Bs[nn + 2][kk] = (__bf16)val.z;
                Bs[nn + 3][kk] = (__bf16)val.w;
            }
        } else {
            #pragma unroll
            for (int t = 0; t < 4; ++t) {
                int linear = tid + t * TB;
                int kk = linear >> 5;
                int nn = (linear & 31) << 2;
                float4 val = make_float4(0.f, 0.f, 0.f, 0.f);
                if ((k0 + kk) < kend)
                    val = *(const float4*)(B + (size_t)(k0 + kk) * ldb + n0 + nn);
                Bs[nn + 0][kk] = (__bf16)val.x;
                Bs[nn + 1][kk] = (__bf16)val.y;
                Bs[nn + 2][kk] = (__bf16)val.z;
                Bs[nn + 3][kk] = (__bf16)val.w;
            }
        }
        __syncthreads();

        // ---- fragments per ISA 7.12.2 layouts (16B-aligned b128 LDS loads) ----
        v16bf afrag[4], bfrag[2];
        #pragma unroll
        for (int i = 0; i < 4; ++i) {
            const __bf16* ap = &As[wm + i * 16 + l16][lhi * 8];
            v8bf lo = *(const v8bf*)ap;             // K = 8*lhi + e
            v8bf hi = *(const v8bf*)(ap + 16);      // K = 16 + 8*lhi + e
            afrag[i] = __builtin_shufflevector(lo, hi,
                0,1,2,3,4,5,6,7,8,9,10,11,12,13,14,15);
        }
        #pragma unroll
        for (int j = 0; j < 2; ++j) {
            const __bf16* bp = &Bs[wn + j * 16 + l16][lhi * 8];
            v8bf lo = *(const v8bf*)bp;             // K = 8*lhi + e
            v8bf hi = *(const v8bf*)(bp + 16);      // K = 16 + 8*lhi + e
            bfrag[j] = __builtin_shufflevector(lo, hi,
                0,1,2,3,4,5,6,7,8,9,10,11,12,13,14,15);
        }

        #pragma unroll
        for (int i = 0; i < 4; ++i)
            #pragma unroll
            for (int j = 0; j < 2; ++j)
                acc[i][j] = __builtin_amdgcn_wmma_f32_16x16x32_bf16(
                    false, afrag[i], false, bfrag[j],
                    (short)0, acc[i][j], false, false);
        __syncthreads();
    }

    // ---- epilogue ----
    if (nchunk > 1) {
        // split-K partial: raw atomic accumulate (no bias/act/scale in this mode)
        #pragma unroll
        for (int i = 0; i < 4; ++i) {
            int rowbase = m0 + wm + i * 16 + lhi * 8;
            #pragma unroll
            for (int j = 0; j < 2; ++j) {
                int col = n0 + wn + j * 16 + l16;
                #pragma unroll
                for (int r = 0; r < 8; ++r) {
                    int row = rowbase + r;
                    if (row < M)
                        atomicAdd(&C[(size_t)row * ldc + col], acc[i][j][r]);
                }
            }
        }
        return;
    }

    #pragma unroll
    for (int i = 0; i < 4; ++i) {
        int rowbase = m0 + wm + i * 16 + lhi * 8;
        #pragma unroll
        for (int j = 0; j < 2; ++j) {
            int col = n0 + wn + j * 16 + l16;
            float bv = bias ? bias[col] : 0.f;
            #pragma unroll
            for (int r = 0; r < 8; ++r) {
                int row = rowbase + r;
                if (mfull || row < M) {
                    float val = acc[i][j][r] + bv;
                    if (beta)    val += C[(size_t)row * ldc + col];
                    if (ACT == 1) val = fmaxf(val, 0.f);
                    else if (ACT == 2) val = (val > 0.f) ? (val + 1.f) : __expf(val);
                    if (rowscale) val *= rowscale[row];
                    C[(size_t)row * ldc + col] = val;
                }
            }
        }
    }
}

// ---- column sum: out[j] += sum_{rows chunk} X[r,j] -------------------------
__global__ __launch_bounds__(TB)
void colsum_kernel(const float* __restrict__ X, float* __restrict__ out,
                   int Nrows, int H, int rowsPerBlock)
{
    int j  = blockIdx.x * TB + threadIdx.x;
    int r0 = blockIdx.y * rowsPerBlock;
    int r1 = min(r0 + rowsPerBlock, Nrows);
    if (j >= H) return;
    float s = 0.f;
    for (int r = r0; r < r1; ++r) s += X[(size_t)r * H + j];
    atomicAdd(&out[j], s);
}

// ---- z[i] = 1/max(q[i,:].ksum, 1e-6) --------------------------------------
__global__ __launch_bounds__(TB)
void rowdot_recip_kernel(const float* __restrict__ Q,
                         const float* __restrict__ ksum,
                         float* __restrict__ z, int H)
{
    __shared__ float red[TB];
    int row = blockIdx.x;
    int tid = threadIdx.x;
    float s = 0.f;
    for (int c = tid; c < H; c += TB) s += Q[(size_t)row * H + c] * ksum[c];
    red[tid] = s;
    __syncthreads();
    for (int o = TB / 2; o > 0; o >>= 1) {
        if (tid < o) red[tid] += red[tid + o];
        __syncthreads();
    }
    if (tid == 0) z[row] = 1.f / fmaxf(red[0], 1e-6f);
}

// ---- hg = LayerNorm(hg + h), H == 512 -------------------------------------
__global__ __launch_bounds__(TB)
void ln_residual_kernel(float* __restrict__ hg, const float* __restrict__ h, int H)
{
    __shared__ float s1[TB], s2[TB];
    int row = blockIdx.x;
    int tid = threadIdx.x;
    size_t base = (size_t)row * H;
    float v0 = hg[base + tid]      + h[base + tid];
    float v1 = hg[base + tid + TB] + h[base + tid + TB];
    s1[tid] = v0 + v1;
    s2[tid] = v0 * v0 + v1 * v1;
    __syncthreads();
    for (int o = TB / 2; o > 0; o >>= 1) {
        if (tid < o) { s1[tid] += s1[tid + o]; s2[tid] += s2[tid + o]; }
        __syncthreads();
    }
    float mean = s1[0] / (float)H;
    float var  = s2[0] / (float)H - mean * mean;
    float inv  = rsqrtf(var + 1e-5f);
    hg[base + tid]      = (v0 - mean) * inv;
    hg[base + tid + TB] = (v1 - mean) * inv;
}

// ---- GCN helpers ----------------------------------------------------------
__global__ void deg_kernel(const int* __restrict__ ei, float* __restrict__ deg, int E)
{
    int e = blockIdx.x * blockDim.x + threadIdx.x;
    if (e < E) atomicAdd(&deg[ei[E + e]], 1.f);      // dst row
}

__global__ void dis_kernel(float* __restrict__ deg, int n)
{
    int i = blockIdx.x * blockDim.x + threadIdx.x;
    if (i < n) deg[i] = rsqrtf(deg[i] + 1.f);        // +1 self-loop; deg>0 always
}

__global__ __launch_bounds__(TB)
void gcn_scatter_kernel(const int* __restrict__ ei, const float* __restrict__ xw,
                        const float* __restrict__ dis, float* __restrict__ acc,
                        int E, int H)
{
    int e = blockIdx.x;
    int s = ei[e];
    int d = ei[E + e];
    float nrm = dis[s] * dis[d];
    for (int c = threadIdx.x; c < H; c += blockDim.x)
        atomicAdd(&acc[(size_t)d * H + c], xw[(size_t)s * H + c] * nrm);
}

__global__ __launch_bounds__(TB)
void gcn_finalize_kernel(const float* __restrict__ acc, const float* __restrict__ xw,
                         const float* __restrict__ dis, const float* __restrict__ b,
                         float* __restrict__ out, int H)
{
    int i = blockIdx.x;
    float selfn = dis[i] * dis[i];
    for (int c = threadIdx.x; c < H; c += blockDim.x) {
        float v = acc[(size_t)i * H + c] + xw[(size_t)i * H + c] * selfn + b[c];
        out[(size_t)i * H + c] = fmaxf(v, 0.f);
    }
}

// ---- host-side dispatch ---------------------------------------------------
static void launch_gemm(int act, bool transA,
                        const float* A, const float* B, float* C,
                        int M, int Nd, int K, int lda, int ldb, int ldc,
                        const float* bias, const float* rowscale, int beta,
                        int splitk, hipStream_t s)
{
    dim3 grid((Nd + BN - 1) / BN, (M + BM - 1) / BM, splitk);
    dim3 blk(TB);
    if (transA) {
        gemm_bf16_wmma<0, true><<<grid, blk, 0, s>>>(A, B, C, M, Nd, K, lda, ldb, ldc, bias, rowscale, beta);
        return;
    }
    switch (act) {
    case 1:  gemm_bf16_wmma<1, false><<<grid, blk, 0, s>>>(A, B, C, M, Nd, K, lda, ldb, ldc, bias, rowscale, beta); break;
    case 2:  gemm_bf16_wmma<2, false><<<grid, blk, 0, s>>>(A, B, C, M, Nd, K, lda, ldb, ldc, bias, rowscale, beta); break;
    default: gemm_bf16_wmma<0, false><<<grid, blk, 0, s>>>(A, B, C, M, Nd, K, lda, ldb, ldc, bias, rowscale, beta); break;
    }
}

extern "C" void kernel_launch(void* const* d_in, const int* in_sizes, int n_in,
                              void* d_out, int out_size, void* d_ws, size_t ws_size,
                              hipStream_t stream)
{
    (void)in_sizes; (void)n_in; (void)out_size; (void)ws_size;

    const float* x     = (const float*)d_in[0];
    const int*   ei    = (const int*)  d_in[1];   // [2,E] (jax default x64-off -> int32)
    const float* W_in  = (const float*)d_in[2];
    const float* b_in  = (const float*)d_in[3];
    const float* Wq    = (const float*)d_in[4];
    const float* Wk    = (const float*)d_in[5];
    const float* Wv    = (const float*)d_in[6];
    const float* Wo    = (const float*)d_in[7];
    const float* bo    = (const float*)d_in[8];
    const float* W_gcn = (const float*)d_in[9];
    const float* b_gcn = (const float*)d_in[10];
    const float* W_mix = (const float*)d_in[11];
    const float* b_mix = (const float*)d_in[12];
    const float* W_cls = (const float*)d_in[13];
    const float* b_cls = (const float*)d_in[14];
    float* out = (float*)d_out;

    constexpr int N = 10000, DIN = 512, H = 512, OUTC = 128, L = 3, E = 160000;
    const size_t NH = (size_t)N * H;
    const size_t HH = (size_t)H * H;

    float* ws   = (float*)d_ws;
    float* x0   = ws;  ws += NH;      // input proj (later reused as hmix)
    float* hg   = ws;  ws += NH;      // h_global
    float* qb   = ws;  ws += NH;      // q / later xw
    float* kb   = ws;  ws += NH;      // k / h1 / later gcn acc
    float* vb   = ws;  ws += NH;      // v / h2 / later h_graph
    float* kv   = ws;  ws += HH;
    float* ksum = ws;  ws += H;
    float* zb   = ws;  ws += N;
    float* deg  = ws;  ws += N;       // degree, then dis (in place)

    // x0 = relu(x @ W_in + b_in)
    launch_gemm(1, false, x, W_in, x0, N, H, DIN, DIN, H, H, b_in, nullptr, 0, 1, stream);
    // h_global = x0
    hipMemcpyAsync(hg, x0, NH * sizeof(float), hipMemcpyDeviceToDevice, stream);

    for (int l = 0; l < L; ++l) {
        const float* Wq_l = Wq + (size_t)l * HH;
        const float* Wk_l = Wk + (size_t)l * HH;
        const float* Wv_l = Wv + (size_t)l * HH;
        const float* Wo_l = Wo + (size_t)l * HH;
        const float* bo_l = bo + (size_t)l * H;

        launch_gemm(2, false, hg, Wq_l, qb, N, H, H, H, H, H, nullptr, nullptr, 0, 1, stream); // q=elu+1
        launch_gemm(2, false, hg, Wk_l, kb, N, H, H, H, H, H, nullptr, nullptr, 0, 1, stream); // k=elu+1
        launch_gemm(0, false, hg, Wv_l, vb, N, H, H, H, H, H, nullptr, nullptr, 0, 1, stream); // v

        // kv = k^T @ v   (M=H, N=H, K=Nrows) -- split-K=16 -> 256 blocks
        hipMemsetAsync(kv, 0, HH * sizeof(float), stream);
        launch_gemm(0, true, kb, vb, kv, H, H, N, H, H, H, nullptr, nullptr, 0, 16, stream);

        // ksum = sum_rows(k)
        hipMemsetAsync(ksum, 0, H * sizeof(float), stream);
        {
            dim3 g(H / TB, 40);
            colsum_kernel<<<g, TB, 0, stream>>>(kb, ksum, N, H, (N + 39) / 40);
        }
        // z = 1/max(q @ ksum, 1e-6)
        rowdot_recip_kernel<<<N, TB, 0, stream>>>(qb, ksum, zb, H);

        // h1 = (q @ kv) * z[:,None]   -> kb (k is dead)
        launch_gemm(0, false, qb, kv, kb, N, H, H, H, H, H, nullptr, zb, 0, 1, stream);
        // h2 = h1 @ Wo + bo           -> vb (v is dead)
        launch_gemm(0, false, kb, Wo_l, vb, N, H, H, H, H, H, bo_l, nullptr, 0, 1, stream);
        // hg = LN(hg + h2)
        ln_residual_kernel<<<N, TB, 0, stream>>>(hg, vb, H);
    }

    // ---- GCN branch: xw = x0 @ W_gcn -> qb ; aggregate into kb ; relu -> vb
    launch_gemm(0, false, x0, W_gcn, qb, N, H, H, H, H, H, nullptr, nullptr, 0, 1, stream);
    hipMemsetAsync(kb, 0, NH * sizeof(float), stream);
    hipMemsetAsync(deg, 0, N * sizeof(float), stream);
    deg_kernel<<<(E + TB - 1) / TB, TB, 0, stream>>>(ei, deg, E);
    dis_kernel<<<(N + TB - 1) / TB, TB, 0, stream>>>(deg, N);
    gcn_scatter_kernel<<<E, TB, 0, stream>>>(ei, qb, deg, kb, E, H);
    gcn_finalize_kernel<<<N, TB, 0, stream>>>(kb, qb, deg, b_gcn, vb, H);

    // ---- mix: hmix = relu(hg @ Wmix_top + h_graph @ Wmix_bot + b_mix) -> x0
    launch_gemm(0, false, hg, W_mix, x0, N, H, H, H, H, H, nullptr, nullptr, 0, 1, stream);
    launch_gemm(1, false, vb, W_mix + HH, x0, N, H, H, H, H, H, b_mix, nullptr, 1, 1, stream);

    // ---- classifier: out = hmix @ W_cls + b_cls
    launch_gemm(0, false, x0, W_cls, out, N, OUTC, H, H, OUTC, OUTC, b_cls, nullptr, 0, 1, stream);
}
